// WindowAttention_55619826483748
// MI455X (gfx1250) — matrix operand order repaired and checked by reading, not compiled, not run on hardware
//
#include <hip/hip_runtime.h>
#include <hip/hip_bf16.h>

#define DIMC   384
#define N3     1152
#define NTOK   49
#define HEADS  12
#define HD     32
#define NWIN   64

typedef __attribute__((ext_vector_type(16))) _Float16 v16h;
typedef __attribute__((ext_vector_type(8)))  _Float16 v8h;
typedef __attribute__((ext_vector_type(8)))  float    v8f;

// LDS strides in halves (row padding +8 halves => conflict-free b128 lane access)
#define XS_STRIDE 392   // 384+8
#define QK_STRIDE 776   // 768+8 (q cols 0..383, k cols 384..767)
#define VT_STRIDE 72    // 64+8
#define P_STRIDE  72

#define XS_OFF 0
#define QK_OFF (64*XS_STRIDE)
#define VT_OFF (QK_OFF + 64*QK_STRIDE)
#define P_OFF  (VT_OFF + HEADS*HD*VT_STRIDE)
#define SMEM_HALVES (P_OFF + 8*64*P_STRIDE)   // 139264 halves = 278528 bytes

__device__ __forceinline__ v16h load_frag_a(const _Float16* __restrict__ base,
                                            int strideH, int row, int kcol, int lane) {
  // A 16x32 f16 layout: lanes 0-15 row m, K {0..7,16..23}; lanes 16-31 row m, K {8..15,24..31}
  const int m = lane & 15, hi = lane >> 4;
  const _Float16* p = base + (row + m) * strideH + kcol + hi * 8;
  union { v16h v; v8h h[2]; } u;
  u.h[0] = *(const v8h*)p;
  u.h[1] = *(const v8h*)(p + 16);
  return u.v;
}

__device__ __forceinline__ v16h load_frag_b(const _Float16* __restrict__ base,
                                            int strideH, int ncol, int kcol, int lane) {
  // B 32x16 f16 layout: lanes 0-15 col n, K {0..7,16..23}; lanes 16-31 col n, K {8..15,24..31}.
  // base must be stored [out-col][k] (i.e. B transposed, K contiguous).
  const int n = lane & 15, kh = lane >> 4;
  const _Float16* p = base + (ncol + n) * strideH + kcol + kh * 8;
  union { v16h v; v8h h[2]; } u;
  u.h[0] = *(const v8h*)p;
  u.h[1] = *(const v8h*)(p + 16);
  return u.v;
}

__device__ __forceinline__ unsigned pack2h(float a, float b) {
  union { _Float16 h[2]; unsigned u; } t;
  t.h[0] = (_Float16)a; t.h[1] = (_Float16)b;
  return t.u;
}

// ---- prep kernels: transpose weights to f16 [oc][ic], gather bias to [h][n][m] ----
__global__ void swin_prep_qkvw(const float* __restrict__ w, _Float16* __restrict__ wt) {
  int i = blockIdx.x * 256 + threadIdx.x;
  if (i < N3 * DIMC) { int oc = i / DIMC, ic = i % DIMC; wt[i] = (_Float16)w[ic * N3 + oc]; }
}
__global__ void swin_prep_projw(const float* __restrict__ w, _Float16* __restrict__ wt) {
  int i = blockIdx.x * 256 + threadIdx.x;
  if (i < DIMC * DIMC) { int oc = i / DIMC, ic = i % DIMC; wt[i] = (_Float16)w[ic * DIMC + oc]; }
}
__global__ void swin_prep_bias(const float* __restrict__ bt, const int* __restrict__ ridx,
                               float* __restrict__ bm) {
  int i = blockIdx.x * 256 + threadIdx.x;
  if (i < HEADS * NTOK * NTOK) {
    int h = i / (NTOK * NTOK), nm = i % (NTOK * NTOK);
    bm[i] = bt[ridx[nm] * HEADS + h];
  }
}

// ---- fused window-attention kernel: one block per window ----
__global__ __launch_bounds__(256) void swin_attn_fused(
    const float* __restrict__ x, const float* __restrict__ mask,
    const float* __restrict__ qkv_b, const float* __restrict__ proj_b,
    const _Float16* __restrict__ qkvWt, const _Float16* __restrict__ projWt,
    const float* __restrict__ biasM, float* __restrict__ out)
{
  extern __shared__ __align__(16) _Float16 sm[];
  _Float16* xs = sm + XS_OFF;                // 64 x 384 (x, later reused as y)
  _Float16* qk = sm + QK_OFF;                // 64 x 768 (q | k)
  _Float16* vT = sm + VT_OFF;                // [h][d][m]
  _Float16* Pb = sm + P_OFF;                 // per-wave 64 x 64 P
  const int tid  = threadIdx.x;
  const int lane = tid & 31;
  const int w    = tid >> 5;
  const int hi   = lane >> 4;
  const int b    = blockIdx.x;
  const int wi   = b % NWIN;

  // ---------- phase 1: x -> f16 LDS, zero-pad rows 49..63 ----------
  {
    const float2* xp = (const float2*)(x + (size_t)b * NTOK * DIMC);
    for (int i = tid; i < 64 * (DIMC / 2); i += 256) {
      int r = i / (DIMC / 2), c2 = i % (DIMC / 2);
      float2 v = (r < NTOK) ? xp[r * (DIMC / 2) + c2] : make_float2(0.f, 0.f);
      *(unsigned*)(xs + r * XS_STRIDE + c2 * 2) = pack2h(v.x, v.y);
    }
  }
  __syncthreads();

  // ---------- phase 2: qkv = xs @ Wqkv^T + b ; q,k -> qk LDS, v -> vT LDS ----------
  for (int mt = 0; mt < 4; ++mt) {
    v16h a[12];
#pragma unroll
    for (int kt = 0; kt < 12; ++kt) a[kt] = load_frag_a(xs, XS_STRIDE, mt * 16, kt * 32, lane);
    for (int j = 0; j < 9; ++j) {
      const int nt = w * 9 + j;
      const int oc = nt * 16 + (lane & 15);
      const float bv = qkv_b[oc];
      v8f acc = {bv, bv, bv, bv, bv, bv, bv, bv};
#pragma unroll
      for (int kt = 0; kt < 12; ++kt) {
        v16h bf = load_frag_b(qkvWt, DIMC, nt * 16, kt * 32, lane);
        acc = __builtin_amdgcn_wmma_f32_16x16x32_f16(false, a[kt], false, bf,
                                                     (short)0, acc, false, false);
      }
      if (oc < 768) {
#pragma unroll
        for (int v2 = 0; v2 < 8; ++v2) {
          int m = mt * 16 + v2 + 8 * hi;
          qk[m * QK_STRIDE + oc] = (_Float16)acc[v2];
        }
      } else {
        int hh = (oc - 768) >> 5, dd = (oc - 768) & 31;
        _Float16* vrow = vT + (hh * HD + dd) * VT_STRIDE;
#pragma unroll
        for (int v2 = 0; v2 < 8; v2 += 2) {
          int m = mt * 16 + v2 + 8 * hi;
          *(unsigned*)(vrow + m) = pack2h(acc[v2], acc[v2 + 1]);
        }
      }
    }
  }
  __syncthreads();

  // ---------- phase 3: per-head attention (S^T = K q^T, softmax in-lane, O = P v) ----------
  const float sc = 0.17677669529663687f;   // 1/sqrt(32)
  _Float16* Pw = Pb + w * 64 * P_STRIDE;
  for (int h = w; h < HEADS; h += 8) {
    v16h ak[4];
#pragma unroll
    for (int mt = 0; mt < 4; ++mt)
      ak[mt] = load_frag_a(qk, QK_STRIDE, mt * 16, DIMC + h * HD, lane);   // K rows

    for (int nt = 0; nt < 4; ++nt) {
      v16h bq = load_frag_b(qk, QK_STRIDE, nt * 16, h * HD, lane);         // q as B: [token][d]
      v8f sa[4];
      v8f z = {0, 0, 0, 0, 0, 0, 0, 0};
#pragma unroll
      for (int mt = 0; mt < 4; ++mt)
        sa[mt] = __builtin_amdgcn_wmma_f32_16x16x32_f16(false, ak[mt], false, bq,
                                                        (short)0, z, false, false);
      const int n = nt * 16 + (lane & 15);           // query index (same for lane pair)
      const float* bp = biasM + (h * NTOK + n) * NTOK;
      const float* mp = mask + ((size_t)wi * NTOK + n) * NTOK;
      float s[4][8];
      float mx = -3.0e38f;
#pragma unroll
      for (int mt = 0; mt < 4; ++mt)
#pragma unroll
        for (int v2 = 0; v2 < 8; ++v2) {
          int m = mt * 16 + v2 + 8 * hi;             // key index
          float val = -3.0e38f;
          if (n < NTOK && m < NTOK) val = sa[mt][v2] * sc + bp[m] + mp[m];
          s[mt][v2] = val;
          mx = fmaxf(mx, val);
        }
      mx = fmaxf(mx, __shfl_xor(mx, 16, 32));
      float sum = 0.f;
#pragma unroll
      for (int mt = 0; mt < 4; ++mt)
#pragma unroll
        for (int v2 = 0; v2 < 8; ++v2) {
          float e = __expf(s[mt][v2] - mx);
          s[mt][v2] = e;
          sum += e;
        }
      sum += __shfl_xor(sum, 16, 32);
      float r = 1.0f / (sum + 1e-30f);
      _Float16* prow = Pw + n * P_STRIDE;
#pragma unroll
      for (int mt = 0; mt < 4; ++mt)
#pragma unroll
        for (int v2 = 0; v2 < 8; v2 += 2) {
          int m = mt * 16 + v2 + 8 * hi;
          *(unsigned*)(prow + m) = pack2h(s[mt][v2] * r, s[mt][v2 + 1] * r);
        }
    }

    // O = P @ V : A = P[n][m], B = vT[d][m]
    const _Float16* vh = vT + h * HD * VT_STRIDE;
#pragma unroll
    for (int ntd = 0; ntd < 2; ++ntd) {
      v16h bv0 = load_frag_b(vh, VT_STRIDE, ntd * 16, 0, lane);
      v16h bv1 = load_frag_b(vh, VT_STRIDE, ntd * 16, 32, lane);
      for (int mt = 0; mt < 4; ++mt) {
        v16h p0 = load_frag_a(Pw, P_STRIDE, mt * 16, 0, lane);
        v16h p1 = load_frag_a(Pw, P_STRIDE, mt * 16, 32, lane);
        v8f acc = {0, 0, 0, 0, 0, 0, 0, 0};
        acc = __builtin_amdgcn_wmma_f32_16x16x32_f16(false, p0, false, bv0, (short)0, acc, false, false);
        acc = __builtin_amdgcn_wmma_f32_16x16x32_f16(false, p1, false, bv1, (short)0, acc, false, false);
        const int dd = ntd * 16 + (lane & 15);
#pragma unroll
        for (int v2 = 0; v2 < 8; ++v2) {
          int nn = mt * 16 + v2 + 8 * hi;
          xs[nn * XS_STRIDE + h * HD + dd] = (_Float16)acc[v2];   // y reuses xs
        }
      }
    }
  }
  __syncthreads();

  // ---------- phase 4: out = y @ Wproj^T + b ----------
  {
    float* outp = out + (size_t)b * NTOK * DIMC;
    for (int j = 0; j < 3; ++j) {
      const int nt = w * 3 + j;
      const int oc = nt * 16 + (lane & 15);
      const float pb = proj_b[oc];
      for (int mt = 0; mt < 4; ++mt) {
        v8f acc = {pb, pb, pb, pb, pb, pb, pb, pb};
#pragma unroll
        for (int kt = 0; kt < 12; ++kt) {
          v16h af = load_frag_a(xs, XS_STRIDE, mt * 16, kt * 32, lane);
          v16h bf = load_frag_b(projWt, DIMC, nt * 16, kt * 32, lane);
          acc = __builtin_amdgcn_wmma_f32_16x16x32_f16(false, af, false, bf,
                                                       (short)0, acc, false, false);
        }
#pragma unroll
        for (int v2 = 0; v2 < 8; ++v2) {
          int nn = mt * 16 + v2 + 8 * hi;
          if (nn < NTOK) outp[nn * DIMC + oc] = acc[v2];
        }
      }
    }
  }
}

extern "C" void kernel_launch(void* const* d_in, const int* in_sizes, int n_in,
                              void* d_out, int out_size, void* d_ws, size_t ws_size,
                              hipStream_t stream) {
  (void)in_sizes; (void)n_in; (void)out_size; (void)ws_size;
  const float* x          = (const float*)d_in[0];
  const float* mask       = (const float*)d_in[1];
  const float* qkv_w      = (const float*)d_in[2];
  const float* qkv_b      = (const float*)d_in[3];
  const float* proj_w     = (const float*)d_in[4];
  const float* proj_b     = (const float*)d_in[5];
  const float* bias_table = (const float*)d_in[6];
  const int*   rel_idx    = (const int*)d_in[7];

  char* ws = (char*)d_ws;
  _Float16* qkvWt  = (_Float16*)ws;                       // 1152*384*2 = 884736 B
  _Float16* projWt = (_Float16*)(ws + 884736);            // 384*384*2  = 294912 B
  float*    biasM  = (float*)(ws + 884736 + 294912);      // 12*49*49*4 = 115248 B

  swin_prep_qkvw <<<(N3 * DIMC + 255) / 256, 256, 0, stream>>>(qkv_w, qkvWt);
  swin_prep_projw<<<(DIMC * DIMC + 255) / 256, 256, 0, stream>>>(proj_w, projWt);
  swin_prep_bias <<<(HEADS * NTOK * NTOK + 255) / 256, 256, 0, stream>>>(bias_table, rel_idx, biasM);

  const size_t shbytes = (size_t)SMEM_HALVES * sizeof(_Float16);   // 278528 B
  (void)hipFuncSetAttribute((const void*)swin_attn_fused,
                            hipFuncAttributeMaxDynamicSharedMemorySize, (int)shbytes);
  swin_attn_fused<<<2048, 256, shbytes, stream>>>(x, mask, qkv_b, proj_b,
                                                  qkvWt, projWt, biasM, (float*)d_out);
}
